// Mamba2LM_83502754169004
// MI455X (gfx1250) — compile-verified, matrix operand order checked by reading, not run on hardware
//
#include <hip/hip_runtime.h>
#include <hip/hip_bf16.h>

// ---------------- model constants ----------------
#define SEQ      1024
#define DMODEL   1024
#define NLAYERS  4
#define DSTATE   64
#define DCONV    4
#define DINNER   2048
#define HEADDIM  64
#define NHEADS   32
#define CONVDIM  2176          // DINNER + 2*DSTATE
#define DINPROJ  4256          // 2*DINNER + 2*DSTATE + NHEADS
#define VOCAB    32000
#define EPS      1e-5f

typedef __attribute__((ext_vector_type(16))) __bf16 v16bf;
typedef __attribute__((ext_vector_type(8)))  float  v8f;

#define WMMA_BF16(A_, B_, C_) \
    __builtin_amdgcn_wmma_f32_16x16x32_bf16(false, (A_), false, (B_), (short)0, (C_), false, false)

__device__ __forceinline__ unsigned short f2bf(float f) {
    unsigned int u = __float_as_uint(f);
    u += 0x7FFFu + ((u >> 16) & 1u);          // round-to-nearest-even
    return (unsigned short)(u >> 16);
}
__device__ __forceinline__ float siluf(float x) {
    return x / (1.0f + __expf(-x));
}

// ---------------- fp32 -> bf16 conversion ----------------
__global__ void k_cvt_bf16(const float* __restrict__ src,
                           unsigned short* __restrict__ dst, int n) {
    int i = blockIdx.x * blockDim.x + threadIdx.x;
    if (i < n) dst[i] = f2bf(src[i]);
}

// ---------------- embedding gather ----------------
__global__ void k_embed(const int* __restrict__ ids,
                        const float* __restrict__ emb,
                        float* __restrict__ x) {
    int i = blockIdx.x * blockDim.x + threadIdx.x;   // SEQ*DMODEL threads
    if (i >= SEQ * DMODEL) return;
    int l = i >> 10, d = i & (DMODEL - 1);
    x[i] = emb[(size_t)ids[l] * DMODEL + d];
}

// ---------------- RMSNorm -> bf16 (one block per row) ----------------
__global__ void k_rmsnorm_bf16(const float* __restrict__ x,
                               const float* __restrict__ w,
                               unsigned short* __restrict__ out, int cols) {
    int row = blockIdx.x;
    const float* xr = x + (size_t)row * cols;
    __shared__ float red[256];
    float ss = 0.f;
    for (int c = threadIdx.x; c < cols; c += 256) { float v = xr[c]; ss += v * v; }
    red[threadIdx.x] = ss; __syncthreads();
    for (int s = 128; s > 0; s >>= 1) {
        if (threadIdx.x < s) red[threadIdx.x] += red[threadIdx.x + s];
        __syncthreads();
    }
    float scale = rsqrtf(red[0] / (float)cols + EPS);
    for (int c = threadIdx.x; c < cols; c += 256)
        out[(size_t)row * cols + c] = f2bf(xr[c] * scale * w[c]);
}

// ---------------- bf16 WMMA GEMM:  C[M,N] (+)= A[M,K] * W[N,K]^T ----------------
// Wave tile: 16x64 output (4 accumulators, one A fragment feeds 4 WMMAs).
// Block: 256 threads = 8 waves -> 128 rows of M.  Grid: (ceil(N/64), M/128).
// Tail blocks (N % 64 == 32, e.g. N=4256) drop to a 16x32 tile.
// Fragment loads are contiguous 32B per lane (k = 16*(lane/16) + s in A and B); the bf16
// weight set (117MB) fits in the 192MB L2, so cross-tile re-reads hit L2.
// global_prefetch warms the weight stream ~1KB ahead (no LOADcnt cost).
__global__ void k_gemm_bf16(const unsigned short* __restrict__ A,
                            const unsigned short* __restrict__ W,
                            float* __restrict__ C,
                            int M, int N, int K, int addmode) {
    int lane = threadIdx.x & 31;
    int wave = threadIdx.x >> 5;
    int m0 = blockIdx.y * 128 + wave * 16;
    int n0 = blockIdx.x * 64;
    if (m0 >= M || n0 >= N) return;

    int r  = lane & 15;
    int kh = (lane >> 4) << 4;                 // 0 or 16: lane-half K offset
    const unsigned short* Arow = A + (size_t)(m0 + r) * K + kh;
    const unsigned short* Wr   = W + (size_t)(n0 + r) * K + kh;
    const size_t ws16 = (size_t)16 * K;        // 16 weight rows
    const bool full = (n0 + 64 <= N);

    v8f acc0 = {}; v8f acc1 = {}; v8f acc2 = {}; v8f acc3 = {};
    if (full) {
        for (int kb = 0; kb < K; kb += 32) {
            __builtin_prefetch(Wr + kb + 512, 0, 1);            // ~1KB ahead
            v16bf a  = *reinterpret_cast<const v16bf*>(Arow + kb);
            v16bf b0 = *reinterpret_cast<const v16bf*>(Wr + kb);
            v16bf b1 = *reinterpret_cast<const v16bf*>(Wr + ws16 + kb);
            v16bf b2 = *reinterpret_cast<const v16bf*>(Wr + 2 * ws16 + kb);
            v16bf b3 = *reinterpret_cast<const v16bf*>(Wr + 3 * ws16 + kb);
            acc0 = WMMA_BF16(a, b0, acc0);
            acc1 = WMMA_BF16(a, b1, acc1);
            acc2 = WMMA_BF16(a, b2, acc2);
            acc3 = WMMA_BF16(a, b3, acc3);
        }
    } else {
        for (int kb = 0; kb < K; kb += 32) {
            v16bf a  = *reinterpret_cast<const v16bf*>(Arow + kb);
            v16bf b0 = *reinterpret_cast<const v16bf*>(Wr + kb);
            v16bf b1 = *reinterpret_cast<const v16bf*>(Wr + ws16 + kb);
            acc0 = WMMA_BF16(a, b0, acc0);
            acc1 = WMMA_BF16(a, b1, acc1);
        }
    }

    // C/D layout: VGPR i -> (M = m0 + i + 8*(lane>>4), N = n0 + (lane&15))
    int mrow = m0 + ((lane >> 4) << 3);
    int ncol = n0 + (lane & 15);
    float* Cp = C + (size_t)mrow * N + ncol;
    if (addmode) {
#pragma unroll
        for (int i = 0; i < 8; ++i) {
            Cp[(size_t)i * N]      += acc0[i];
            Cp[(size_t)i * N + 16] += acc1[i];
        }
        if (full) {
#pragma unroll
            for (int i = 0; i < 8; ++i) {
                Cp[(size_t)i * N + 32] += acc2[i];
                Cp[(size_t)i * N + 48] += acc3[i];
            }
        }
    } else {
#pragma unroll
        for (int i = 0; i < 8; ++i) {
            Cp[(size_t)i * N]      = acc0[i];
            Cp[(size_t)i * N + 16] = acc1[i];
        }
        if (full) {
#pragma unroll
            for (int i = 0; i < 8; ++i) {
                Cp[(size_t)i * N + 32] = acc2[i];
                Cp[(size_t)i * N + 48] = acc3[i];
            }
        }
    }
}

// ---------------- depthwise causal conv1d + SiLU ----------------
// input = zxbcdt columns [DINNER, DINNER+CONVDIM)
__global__ void k_conv_silu(const float* __restrict__ zx,
                            const float* __restrict__ cw,
                            const float* __restrict__ cb,
                            float* __restrict__ xBC) {
    int i = blockIdx.x * blockDim.x + threadIdx.x;   // SEQ*CONVDIM threads
    if (i >= SEQ * CONVDIM) return;
    int l = i / CONVDIM, c = i - l * CONVDIM;
    float acc = cb[c];
#pragma unroll
    for (int k = 0; k < DCONV; ++k) {
        int t = l + k - (DCONV - 1);
        if (t >= 0) acc += zx[(size_t)t * DINPROJ + DINNER + c] * cw[c * DCONV + k];
    }
    xBC[i] = siluf(acc);
}

// ---------------- dt = softplus(dt_raw + bias) ----------------
__global__ void k_dt(const float* __restrict__ zx,
                     const float* __restrict__ dtb,
                     float* __restrict__ dt) {
    int i = blockIdx.x * blockDim.x + threadIdx.x;   // SEQ*NHEADS
    if (i >= SEQ * NHEADS) return;
    int l = i >> 5, h = i & 31;
    float v = zx[(size_t)l * DINPROJ + (DINNER + CONVDIM) + h] + dtb[h];
    dt[i] = (v > 20.f) ? v : log1pf(__expf(v));
}

// ---------------- selective scan ----------------
// One block per head (32 blocks), 64 threads (= 2 waves); thread p holds
// state row h[p][0..63] in 64 VGPRs.  B/C vectors staged through LDS.
__global__ void k_scan(const float* __restrict__ xBC,
                       const float* __restrict__ dt,
                       const float* __restrict__ A_log,
                       const float* __restrict__ Dskip,
                       float* __restrict__ y) {
    int head = blockIdx.x;
    int p    = threadIdx.x;                  // 0..63
    float h[DSTATE];
#pragma unroll
    for (int n = 0; n < DSTATE; ++n) h[n] = 0.f;
    float Ah = -__expf(A_log[head]);
    float Dh = Dskip[head];
    __shared__ float sB[DSTATE], sC[DSTATE];
    for (int t = 0; t < SEQ; ++t) {
        const float* row = xBC + (size_t)t * CONVDIM;
        sB[p] = row[DINNER + p];
        sC[p] = row[DINNER + DSTATE + p];
        __syncthreads();
        float dtt   = dt[t * NHEADS + head];
        float dA    = __expf(dtt * Ah);
        float xt    = row[head * HEADDIM + p];
        float scale = dtt * xt;
        float acc = 0.f;
#pragma unroll
        for (int n = 0; n < DSTATE; ++n) {
            h[n] = h[n] * dA + scale * sB[n];
            acc += h[n] * sC[n];
        }
        y[(size_t)t * DINNER + head * HEADDIM + p] = acc + Dh * xt;
        __syncthreads();
    }
}

// ---------------- gated RMSNorm: rmsnorm(y * silu(z), gw) -> bf16 ----------------
__global__ void k_gate_rmsnorm(const float* __restrict__ y,
                               const float* __restrict__ zx,
                               const float* __restrict__ gw,
                               unsigned short* __restrict__ out) {
    int row = blockIdx.x;
    const float* yr = y  + (size_t)row * DINNER;
    const float* zr = zx + (size_t)row * DINPROJ;   // z = first DINNER cols
    __shared__ float red[256];
    float ss = 0.f;
    for (int c = threadIdx.x; c < DINNER; c += 256) {
        float g = yr[c] * siluf(zr[c]);
        ss += g * g;
    }
    red[threadIdx.x] = ss; __syncthreads();
    for (int s = 128; s > 0; s >>= 1) {
        if (threadIdx.x < s) red[threadIdx.x] += red[threadIdx.x + s];
        __syncthreads();
    }
    float scale = rsqrtf(red[0] / (float)DINNER + EPS);
    for (int c = threadIdx.x; c < DINNER; c += 256) {
        float g = yr[c] * siluf(zr[c]);
        out[(size_t)row * DINNER + c] = f2bf(g * scale * gw[c]);
    }
}

// ---------------- f32 copy ----------------
__global__ void k_copy(const float* __restrict__ src, float* __restrict__ dst, int n) {
    int i = blockIdx.x * blockDim.x + threadIdx.x;
    if (i < n) dst[i] = src[i];
}

// =====================================================================
extern "C" void kernel_launch(void* const* d_in, const int* in_sizes, int n_in,
                              void* d_out, int out_size, void* d_ws, size_t ws_size,
                              hipStream_t stream) {
    const int*   ids       = (const int*)  d_in[0];
    const float* embed     = (const float*)d_in[1];
    const float* norm_w    = (const float*)d_in[2];
    const float* in_proj_w = (const float*)d_in[3];
    const float* conv_w    = (const float*)d_in[4];
    const float* conv_b    = (const float*)d_in[5];
    const float* dt_bias   = (const float*)d_in[6];
    const float* A_log     = (const float*)d_in[7];
    const float* D_skip    = (const float*)d_in[8];
    const float* gnorm_w   = (const float*)d_in[9];
    const float* out_pw    = (const float*)d_in[10];
    const float* fnorm_w   = (const float*)d_in[11];
    const float* lm_w      = (const float*)d_in[12];
    float* out = (float*)d_out;

    // ---- carve workspace (256B aligned) ----
    char* ws = (char*)d_ws;
    size_t off = 0;
    auto carve = [&](size_t bytes) -> char* {
        char* p = ws + off;
        off = (off + bytes + 255) & ~(size_t)255;
        return p;
    };
    const int nWin  = NLAYERS * DINPROJ * DMODEL;   // 17,432,576
    const int nWout = NLAYERS * DMODEL * DINNER;    //  8,388,608
    const int nWlm  = VOCAB * DMODEL;               // 32,768,000
    unsigned short* wInB  = (unsigned short*)carve((size_t)nWin  * 2);
    unsigned short* wOutB = (unsigned short*)carve((size_t)nWout * 2);
    unsigned short* wLmB  = (unsigned short*)carve((size_t)nWlm  * 2);
    float*          x     = (float*)         carve((size_t)SEQ * DMODEL * 4);
    unsigned short* xnB   = (unsigned short*)carve((size_t)SEQ * DMODEL * 2);
    float*          zx    = (float*)         carve((size_t)SEQ * DINPROJ * 4);
    float*          xBC   = (float*)         carve((size_t)SEQ * CONVDIM * 4);
    float*          dtb   = (float*)         carve((size_t)SEQ * NHEADS * 4);
    float*          ybuf  = (float*)         carve((size_t)SEQ * DINNER * 4);
    unsigned short* gB    = (unsigned short*)carve((size_t)SEQ * DINNER * 2);
    unsigned short* hnB   = (unsigned short*)carve((size_t)SEQ * DMODEL * 2);
    (void)ws_size; (void)n_in; (void)in_sizes; (void)out_size;

    // ---- weights -> bf16 (once per launch; deterministic) ----
    k_cvt_bf16<<<(nWin  + 255) / 256, 256, 0, stream>>>(in_proj_w, wInB,  nWin);
    k_cvt_bf16<<<(nWout + 255) / 256, 256, 0, stream>>>(out_pw,    wOutB, nWout);
    k_cvt_bf16<<<(nWlm  + 255) / 256, 256, 0, stream>>>(lm_w,      wLmB,  nWlm);

    // ---- embedding ----
    k_embed<<<(SEQ * DMODEL + 255) / 256, 256, 0, stream>>>(ids, embed, x);

    // ---- layers ----
    for (int i = 0; i < NLAYERS; ++i) {
        k_rmsnorm_bf16<<<SEQ, 256, 0, stream>>>(x, norm_w + i * DMODEL, xnB, DMODEL);

        // zxbcdt = xn @ in_proj_w^T   (M=1024, N=4256, K=1024)
        k_gemm_bf16<<<dim3((DINPROJ + 63) / 64, SEQ / 128), 256, 0, stream>>>(
            xnB, wInB + (size_t)i * DINPROJ * DMODEL, zx,
            SEQ, DINPROJ, DMODEL, 0);

        k_conv_silu<<<(SEQ * CONVDIM + 255) / 256, 256, 0, stream>>>(
            zx, conv_w + (size_t)i * CONVDIM * DCONV, conv_b + (size_t)i * CONVDIM, xBC);

        k_dt<<<(SEQ * NHEADS + 255) / 256, 256, 0, stream>>>(
            zx, dt_bias + i * NHEADS, dtb);

        k_scan<<<NHEADS, HEADDIM, 0, stream>>>(
            xBC, dtb, A_log + i * NHEADS, D_skip + i * NHEADS, ybuf);

        k_gate_rmsnorm<<<SEQ, 256, 0, stream>>>(
            ybuf, zx, gnorm_w + (size_t)i * DINNER, gB);

        // x += gated @ out_proj_w^T   (M=1024, N=1024, K=2048)
        k_gemm_bf16<<<dim3(DMODEL / 64, SEQ / 128), 256, 0, stream>>>(
            gB, wOutB + (size_t)i * DMODEL * DINNER, x,
            SEQ, DMODEL, DINNER, 1);
    }

    // ---- head ----
    k_rmsnorm_bf16<<<SEQ, 256, 0, stream>>>(x, fnorm_w, hnB, DMODEL);

    // logits = hn @ lm_head_w^T   (M=1024, N=32000, K=1024) -> d_out[0:32768000)
    k_gemm_bf16<<<dim3(VOCAB / 64, SEQ / 128), 256, 0, stream>>>(
        hnB, wLmB, out, SEQ, VOCAB, DMODEL, 0);

    // hidden (pre-final-norm x) -> d_out[32768000 : +1048576)
    k_copy<<<(SEQ * DMODEL + 255) / 256, 256, 0, stream>>>(
        x, out + (size_t)SEQ * VOCAB, SEQ * DMODEL);
}
